// Attention_78675210928761
// MI455X (gfx1250) — matrix operand order repaired and checked by reading, not compiled, not run on hardware
//
#include <hip/hip_runtime.h>
#include <hip/hip_bf16.h>

typedef __attribute__((ext_vector_type(16))) _Float16 v16h;
typedef __attribute__((ext_vector_type(8)))  _Float16 v8h;
typedef __attribute__((ext_vector_type(8)))  float    v8f;

#define D_MODEL 192
#define N_HEADS 6
#define HEAD_DIM 32
#define SEQ 2048
#define BATCH 4
#define ROWS (BATCH * SEQ) /* 8192 */

// 1/sqrt(32) * log2(e): base-2 softmax, exp2f lowers to a single v_exp_f32
#define Q_SCALE 0.25506325117567731f

__device__ __forceinline__ v8f wmma_f16(v16h a, v16h b, v8f c) {
  // D = A(16x32 f16) * B(32x16 f16) + C(16x16 f32)
  return __builtin_amdgcn_wmma_f32_16x16x32_f16(false, a, false, b, (short)0, c,
                                                false, false);
}

__device__ __forceinline__ float group16_max(float v) {
  v = fmaxf(v, __shfl_xor(v, 1, 32));
  v = fmaxf(v, __shfl_xor(v, 2, 32));
  v = fmaxf(v, __shfl_xor(v, 4, 32));
  v = fmaxf(v, __shfl_xor(v, 8, 32));
  return v;
}
__device__ __forceinline__ float group16_sum(float v) {
  v += __shfl_xor(v, 1, 32);
  v += __shfl_xor(v, 2, 32);
  v += __shfl_xor(v, 4, 32);
  v += __shfl_xor(v, 8, 32);
  return v;
}

// ---------------------------------------------------------------------------
// Kernel 1: cast X to f16; transpose Wq/Wk/Wv into f16 [din][dout] (B-operand)
// ---------------------------------------------------------------------------
__global__ void prep_kernel(const float* __restrict__ X,
                            const float* __restrict__ Wq,
                            const float* __restrict__ Wk,
                            const float* __restrict__ Wv,
                            _Float16* __restrict__ Xh,
                            _Float16* __restrict__ WqT,
                            _Float16* __restrict__ WkT,
                            _Float16* __restrict__ WvT) {
  int i = blockIdx.x * blockDim.x + threadIdx.x;
  if (i < ROWS * D_MODEL) Xh[i] = (_Float16)X[i];
  if (i < 3 * D_MODEL * D_MODEL) {
    int m = i / (D_MODEL * D_MODEL);
    int rem = i - m * (D_MODEL * D_MODEL);
    int r = rem / D_MODEL;        // dout
    int c = rem - r * D_MODEL;    // din
    const float* W = (m == 0) ? Wq : (m == 1) ? Wk : Wv;
    _Float16* T = (m == 0) ? WqT : (m == 1) ? WkT : WvT;
    T[c * D_MODEL + r] = (_Float16)W[r * D_MODEL + c];
  }
}

// ---------------------------------------------------------------------------
// Kernel 2: QKV projection GEMM, one wave per 16x16 output tile, WMMA K-loop.
//   Q -> Qh [B][H][S][32] f16 (pre-scaled by log2e/sqrt(32))
//   K -> KT [B][H][32][S] f16 (transposed: ready as scores B-operand)
//   V -> Vh [B][H][S][32] f16 (ready as context B-operand)
// ---------------------------------------------------------------------------
__global__ void qkv_gemm_kernel(const _Float16* __restrict__ Xh,
                                const _Float16* __restrict__ WqT,
                                const _Float16* __restrict__ WkT,
                                const _Float16* __restrict__ WvT,
                                const float* __restrict__ bq,
                                const float* __restrict__ bk,
                                const float* __restrict__ bv,
                                _Float16* __restrict__ Qh,
                                _Float16* __restrict__ KT,
                                _Float16* __restrict__ Vh) {
  const int l = threadIdx.x;                // lane 0..31
  const int m0 = blockIdx.x * 16;
  const int n0 = blockIdx.y * 16;
  const int mat = blockIdx.z;               // 0=Q 1=K 2=V
  const _Float16* WT = (mat == 0) ? WqT : (mat == 1) ? WkT : WvT;
  const float* bias = (mat == 0) ? bq : (mat == 1) ? bk : bv;

  const int rl = l & 15;                    // A-operand row = lane%16
  const int off = (l < 16) ? 0 : 8;         // A half-K base (ISA 7.12.2)
  const int arow = m0 + rl;

  v8f c;
#pragma unroll
  for (int i = 0; i < 8; ++i) c[i] = 0.0f;

#pragma unroll
  for (int kk = 0; kk < 6; ++kk) {          // K = 192 = 6 * 32
    const int k0 = kk * 32;
    const _Float16* ap = Xh + arow * D_MODEL + k0;
    v8h alo = *(const v8h*)(ap + off);       // K = off .. off+7
    v8h ahi = *(const v8h*)(ap + off + 16);  // K = off+16 .. off+23
    v16h a;
#pragma unroll
    for (int i = 0; i < 8; ++i) { a[i] = alo[i]; a[8 + i] = ahi[i]; }
    // B: lane l <-> N-col (n0+l%16); K in same order as A
    v16h bmat;
#pragma unroll
    for (int i = 0; i < 8; ++i) {
      bmat[i] = WT[(k0 + off + i) * D_MODEL + n0 + rl];
      bmat[8 + i] = WT[(k0 + off + 16 + i) * D_MODEL + n0 + rl];
    }
    c = wmma_f16(a, bmat, c);
  }

  const int n = l & 15;                     // C layout: col = lane%16
  const int col = n0 + n;
  const int h = col >> 5, d = col & 31;
  const float bs = bias[col];
#pragma unroll
  for (int i = 0; i < 8; ++i) {             // C layout: rows i+off
    int r = m0 + i + off;
    int bb = r >> 11, s = r & 2047;         // SEQ = 2048
    float v = c[i] + bs;
    if (mat == 0)
      Qh[(((size_t)bb * N_HEADS + h) * SEQ + s) * HEAD_DIM + d] =
          (_Float16)(v * Q_SCALE);
    else if (mat == 1)
      KT[(((size_t)bb * N_HEADS + h) * HEAD_DIM + d) * SEQ + s] = (_Float16)v;
    else
      Vh[(((size_t)bb * N_HEADS + h) * SEQ + s) * HEAD_DIM + d] = (_Float16)v;
  }
}

// ---------------------------------------------------------------------------
// Kernel 3: flash attention. One wave per (b, h, 16-query tile).
// 64 keys per iteration: 8 WMMAs + one online-softmax update (fixed per-row
// reduction cost amortized over 64 keys). Scores never touch HBM.
// ---------------------------------------------------------------------------
__global__ void flash_attn_kernel(const _Float16* __restrict__ Qh,
                                  const _Float16* __restrict__ KT,
                                  const _Float16* __restrict__ Vh,
                                  float* __restrict__ ctx) {
  const int l = threadIdx.x;
  const int q0 = blockIdx.x * 16;
  const int h = blockIdx.y;
  const int b = blockIdx.z;
  const int rl = l & 15;
  const int off = (l < 16) ? 0 : 8;

  // P tile as f16; row stride 72 halves (144B, dword stride 36) -> the 16
  // row base addresses land on 16 distinct banks for the b128 A-reloads.
  __shared__ __align__(16) _Float16 P16[16][72];

  // Persistent Q A-operand (16 queries x 32 dims, pre-scaled)
  const _Float16* Qp = Qh + (((size_t)b * N_HEADS + h) * SEQ + q0) * HEAD_DIM;
  v8h qlo = *(const v8h*)(Qp + rl * HEAD_DIM + off);
  v8h qhi = *(const v8h*)(Qp + rl * HEAD_DIM + off + 16);
  v16h qa;
#pragma unroll
  for (int i = 0; i < 8; ++i) { qa[i] = qlo[i]; qa[8 + i] = qhi[i]; }

  // Scores B-operand source: Kt row for dim=l (contiguous keys)
  const _Float16* KTl = KT + (((size_t)b * N_HEADS + h) * HEAD_DIM + l) * SEQ;
  const _Float16* Vb = Vh + ((size_t)b * N_HEADS + h) * SEQ * HEAD_DIM;

  v8f acc0, acc1, zero;
  float mrun[8], lrun[8];
#pragma unroll
  for (int i = 0; i < 8; ++i) {
    acc0[i] = 0.0f; acc1[i] = 0.0f; zero[i] = 0.0f;
    mrun[i] = -1e30f; lrun[i] = 0.0f;
  }

  const int NIT = SEQ / 64;
  for (int kt = 0; kt < NIT; ++kt) {
    const int key0 = kt * 64;

    // Prefetch next-next chunk of K^T and V into cache (global_prefetch_b8)
    if (kt + 2 < NIT) {
      __builtin_prefetch((const void*)(KTl + key0 + 128), 0, 0);
      __builtin_prefetch((const void*)(Vb + (size_t)(key0 + 128 + l) * HEAD_DIM),
                         0, 0);
    }

    v16h kb0, kb1, kb2, kb3;
#pragma unroll
    for (int i = 0; i < 8; ++i) {
      const _Float16* r0 = KTl + (ptrdiff_t)(off + i - l) * SEQ + key0 + rl;
      const _Float16* r1 = r0 + 16 * SEQ;
      kb0[i] = r0[0];  kb0[8 + i] = r1[0];
      kb1[i] = r0[16]; kb1[8 + i] = r1[16];
      kb2[i] = r0[32]; kb2[8 + i] = r1[32];
      kb3[i] = r0[48]; kb3[8 + i] = r1[48];
    }
    v8f s0 = wmma_f16(qa, kb0, zero);
    v8f s1 = wmma_f16(qa, kb1, zero);
    v8f s2 = wmma_f16(qa, kb2, zero);
    v8f s3 = wmma_f16(qa, kb3, zero);

    // Online softmax per accumulator row (row = i+off, col = lane%16)
#pragma unroll
    for (int i = 0; i < 8; ++i) {
      float t = fmaxf(fmaxf(s0[i], s1[i]), fmaxf(s2[i], s3[i]));
      t = group16_max(t);
      float mnew = fmaxf(mrun[i], t);
      float p0 = exp2f(s0[i] - mnew);
      float p1 = exp2f(s1[i] - mnew);
      float p2 = exp2f(s2[i] - mnew);
      float p3 = exp2f(s3[i] - mnew);
      float rs = group16_sum((p0 + p1) + (p2 + p3));
      float alpha = exp2f(mrun[i] - mnew);
      lrun[i] = lrun[i] * alpha + rs;
      mrun[i] = mnew;
      acc0[i] *= alpha;
      acc1[i] *= alpha;
      _Float16* pr = &P16[i + off][0];
      pr[rl]      = (_Float16)p0;
      pr[16 + rl] = (_Float16)p1;
      pr[32 + rl] = (_Float16)p2;
      pr[48 + rl] = (_Float16)p3;
    }
    __syncthreads();

    // Re-layout P (C layout in LDS) into two A operands (keys 0..31, 32..63)
    const _Float16* pr = &P16[rl][0];
    v8h pl0 = *(const v8h*)(pr + off);
    v8h ph0 = *(const v8h*)(pr + off + 16);
    v8h pl1 = *(const v8h*)(pr + off + 32);
    v8h ph1 = *(const v8h*)(pr + off + 48);
    v16h pa0, pa1;
#pragma unroll
    for (int i = 0; i < 8; ++i) {
      pa0[i] = pl0[i]; pa0[8 + i] = ph0[i];
      pa1[i] = pl1[i]; pa1[8 + i] = ph1[i];
    }
    __syncthreads();

    // Context: B = V chunk, lane l <-> dim col (l%16), K = keys
    const _Float16* vp0 = Vb + (size_t)key0 * HEAD_DIM + rl;
    const _Float16* vp1 = vp0 + 32 * HEAD_DIM;
    v16h vb00, vb01, vb10, vb11;
#pragma unroll
    for (int i = 0; i < 8; ++i) {
      const int ka = (off + i) * HEAD_DIM, kb = (off + 16 + i) * HEAD_DIM;
      vb00[i] = vp0[ka];      vb00[8 + i] = vp0[kb];
      vb01[i] = vp0[ka + 16]; vb01[8 + i] = vp0[kb + 16];
      vb10[i] = vp1[ka];      vb10[8 + i] = vp1[kb];
      vb11[i] = vp1[ka + 16]; vb11[8 + i] = vp1[kb + 16];
    }
    acc0 = wmma_f16(pa0, vb00, acc0);
    acc0 = wmma_f16(pa1, vb10, acc0);
    acc1 = wmma_f16(pa0, vb01, acc1);
    acc1 = wmma_f16(pa1, vb11, acc1);
  }

  // Normalize, NaN guard, write merged-head context [B][S][192]
#pragma unroll
  for (int i = 0; i < 8; ++i) {
    int s = q0 + i + off;
    float inv = (lrun[i] > 0.0f) ? 1.0f / lrun[i] : 0.0f;
    float o0 = acc0[i] * inv;
    float o1 = acc1[i] * inv;
    if (o0 != o0) o0 = 0.0f;
    if (o1 != o1) o1 = 0.0f;
    float* orow = ctx + ((size_t)b * SEQ + s) * D_MODEL + h * HEAD_DIM;
    orow[rl] = o0;
    orow[16 + rl] = o1;
  }
}

// ---------------------------------------------------------------------------
// Kernel 4: fused  out = LayerNorm(resid + in @ W^T + bias)   (one row/block)
// ---------------------------------------------------------------------------
__global__ void proj_ln_kernel(const float* __restrict__ in,
                               const float* __restrict__ resid,
                               const float* __restrict__ W,
                               const float* __restrict__ bias,
                               const float* __restrict__ g,
                               const float* __restrict__ bln,
                               float* __restrict__ out) {
  __shared__ __align__(16) float xin[D_MODEL];
  __shared__ float red[12];
  const int row = blockIdx.x, t = threadIdx.x;
  xin[t] = in[(size_t)row * D_MODEL + t];
  __syncthreads();

  const float4* w4 = (const float4*)(W + (size_t)t * D_MODEL);
  const float4* x4 = (const float4*)xin;
  float acc = bias[t];
#pragma unroll 8
  for (int k = 0; k < D_MODEL / 4; ++k) {
    float4 wv = w4[k];
    float4 xv = x4[k];
    acc += wv.x * xv.x + wv.y * xv.y + wv.z * xv.z + wv.w * xv.w;
  }
  float val = resid[(size_t)row * D_MODEL + t] + acc;

  float s1 = val, s2 = val * val;
#pragma unroll
  for (int m = 1; m < 32; m <<= 1) {
    s1 += __shfl_xor(s1, m, 32);
    s2 += __shfl_xor(s2, m, 32);
  }
  int w = t >> 5;
  if ((t & 31) == 0) { red[w] = s1; red[6 + w] = s2; }
  __syncthreads();
  float S1 = 0.0f, S2 = 0.0f;
#pragma unroll
  for (int i = 0; i < 6; ++i) { S1 += red[i]; S2 += red[6 + i]; }
  float mu = S1 * (1.0f / D_MODEL);
  float var = S2 * (1.0f / D_MODEL) - mu * mu;
  float y = (val - mu) * rsqrtf(var + 1e-5f) * g[t] + bln[t];
  out[(size_t)row * D_MODEL + t] = y;
}

// ---------------------------------------------------------------------------
extern "C" void kernel_launch(void* const* d_in, const int* in_sizes, int n_in,
                              void* d_out, int out_size, void* d_ws,
                              size_t ws_size, hipStream_t stream) {
  const float* X  = (const float*)d_in[0];
  const float* Wq = (const float*)d_in[1];
  const float* bq = (const float*)d_in[2];
  const float* Wk = (const float*)d_in[3];
  const float* bk = (const float*)d_in[4];
  const float* Wv = (const float*)d_in[5];
  const float* bv = (const float*)d_in[6];
  const float* W3 = (const float*)d_in[7];
  const float* b3 = (const float*)d_in[8];
  const float* W1 = (const float*)d_in[9];
  const float* b1 = (const float*)d_in[10];
  const float* lg = (const float*)d_in[11];
  const float* lb = (const float*)d_in[12];
  float* out = (float*)d_out;

  char* base = (char*)d_ws;
  size_t o = 0;
  auto take = [&](size_t bytes) -> char* {
    char* p = base + o;
    o += (bytes + 255) & ~(size_t)255;
    return p;
  };
  _Float16* Xh  = (_Float16*)take((size_t)ROWS * D_MODEL * 2);
  _Float16* WqT = (_Float16*)take((size_t)D_MODEL * D_MODEL * 2);
  _Float16* WkT = (_Float16*)take((size_t)D_MODEL * D_MODEL * 2);
  _Float16* WvT = (_Float16*)take((size_t)D_MODEL * D_MODEL * 2);
  _Float16* Qh  = (_Float16*)take((size_t)BATCH * N_HEADS * SEQ * HEAD_DIM * 2);
  _Float16* KTd = (_Float16*)take((size_t)BATCH * N_HEADS * HEAD_DIM * SEQ * 2);
  _Float16* Vh  = (_Float16*)take((size_t)BATCH * N_HEADS * SEQ * HEAD_DIM * 2);
  float* ctx    = (float*)take((size_t)ROWS * D_MODEL * 4);
  float* encOut = (float*)take((size_t)ROWS * D_MODEL * 4);
  (void)ws_size; (void)in_sizes; (void)n_in; (void)out_size;

  // 1) f16 casts + weight transposes
  int nconv = ROWS * D_MODEL;
  prep_kernel<<<(nconv + 255) / 256, 256, 0, stream>>>(X, Wq, Wk, Wv, Xh, WqT,
                                                       WkT, WvT);
  // 2) QKV projections via WMMA
  qkv_gemm_kernel<<<dim3(ROWS / 16, D_MODEL / 16, 3), 32, 0, stream>>>(
      Xh, WqT, WkT, WvT, bq, bk, bv, Qh, KTd, Vh);
  // 3) flash attention via WMMA
  flash_attn_kernel<<<dim3(SEQ / 16, N_HEADS, BATCH), 32, 0, stream>>>(
      Qh, KTd, Vh, ctx);
  // 4) out-proj + residual + LN
  proj_ln_kernel<<<ROWS, D_MODEL, 0, stream>>>(ctx, X, W3, b3, lg, lb, encOut);
  // 5) FFN + residual + LN -> final output
  proj_ln_kernel<<<ROWS, D_MODEL, 0, stream>>>(encOut, encOut, W1, b1, lg, lb,
                                               out);
}
